// Discriminator2_56358560858129
// MI455X (gfx1250) — matrix operand order repaired and checked by reading, not compiled, not run on hardware
//
#include <hip/hip_runtime.h>
#include <hip/hip_bf16.h>
#include <math.h>

#define FDIM 268
#define F4   67        // 268 / 4
#define EPS  1e-3f

typedef __attribute__((ext_vector_type(16))) _Float16 v16h;
typedef __attribute__((ext_vector_type(8)))  _Float16 v8h;
typedef __attribute__((ext_vector_type(8)))  float    v8f;

// ---------------- init: deg=1 (self loop), out1=0, out2=0 ----------------
__global__ void k_init(float* deg, float* out1, float* out2, int N, long long NH) {
    long long i = (long long)blockIdx.x * blockDim.x + threadIdx.x;
    if (i < NH) out1[i] = 0.0f;
    if (i < N) { deg[i] = 1.0f; out2[i] = 0.0f; }
}

// ---------------- degree over col ----------------
__global__ void k_degree(const int* __restrict__ col, float* deg, int E) {
    int i = blockIdx.x * blockDim.x + threadIdx.x;
    if (i < E) atomicAdd(&deg[col[i]], 1.0f);
}

// ---------------- dis = rsqrt(max(deg,1)) ----------------
__global__ void k_dis(const float* __restrict__ deg, float* dis, int N) {
    int i = blockIdx.x * blockDim.x + threadIdx.x;
    if (i < N) {
        float d = deg[i];
        dis[i] = (d > 0.0f) ? rsqrtf(fmaxf(d, 1.0f)) : 0.0f;
    }
}

// ---------------- WMMA tile step: fragments from LDS, one 16x16x32 MAC ----------
__device__ __forceinline__ v8f wmma_step(const _Float16* lA, const _Float16* lB,
                                         int wave, int lane, v8f acc) {
    const int m  = lane & 15;
    const int hi = lane >> 4;
    // A (16x32 f16): halves 0..7 -> K=hi*8+j ; halves 8..15 -> K=16+hi*8+j
    const v8h* pa = reinterpret_cast<const v8h*>(&lA[(wave * 16 + m) * 32 + hi * 8]);
    v8h alo = pa[0];
    v8h ahi = pa[2];                       // +16 halves
    // B (32x16 f16): n = lane&15, halves 0..7 -> K=hi*8+j ; halves 8..15 -> K=16+hi*8+j
    const v8h* pb = reinterpret_cast<const v8h*>(&lB[m * 32 + hi * 8]);
    v8h blo = pb[0];
    v8h bhi = pb[2];
    v16h a, b;
    #pragma unroll
    for (int j = 0; j < 8; ++j) {
        a[j] = alo[j]; a[8 + j] = ahi[j];
        b[j] = blo[j]; b[8 + j] = bhi[j];
    }
    return __builtin_amdgcn_wmma_f32_16x16x32_f16(false, a, false, b,
                                                  (short)0, acc, false, false);
}

// ---------------- H = X @ W1 via v_wmma_f32_16x16x32_f16 ----------------
// Block: 256 threads = 8 waves. Block tile: 128 rows x 16 cols.
// K = 268 peeled as 8 guard-free chunks of 32 + one guarded tail (12 valid).
__global__ __launch_bounds__(256) void k_gemm1(const float* __restrict__ X,
                                               const float* __restrict__ W,
                                               float* __restrict__ H, int M) {
    __shared__ _Float16 lA[128 * 32];   // [m][k], row stride 32 halves (64B)
    __shared__ _Float16 lB[16 * 32];    // transposed: [n][k], row stride 32 halves

    const int tid  = threadIdx.x;
    const int lane = tid & 31;
    const int wave = tid >> 5;
    const int row0 = blockIdx.y * 128;
    const int n0   = blockIdx.x * 16;

    // A staging: thread t owns row (t>>1), halves [(t&1)*16, +16)
    const int am = tid >> 1;
    const int ak = (tid & 1) * 16;
    const int gr = row0 + am;
    const bool rowOK = (gr < M);
    const float* xrow = X + (long long)gr * FDIM;
    v8h* aDst = reinterpret_cast<v8h*>(&lA[am * 32 + ak]);

    // B staging: thread t owns k = t>>3, n = (t&7)*2 (+1)  -> coalesced W reads
    const int bk = tid >> 3;
    const int bn = (tid & 7) * 2;
    const int gn = n0 + bn;

    v8f acc = {};

    #pragma unroll
    for (int t = 0; t < 8; ++t) {
        const int k0 = t * 32;
        __syncthreads();
        // ---- stage A: 4x float4 loads -> 2x v8h LDS stores ----
        {
            float f[16];
            if (rowOK) {
                const float4* p = reinterpret_cast<const float4*>(xrow + k0 + ak);
                float4 q0 = p[0], q1 = p[1], q2 = p[2], q3 = p[3];
                f[0]=q0.x; f[1]=q0.y; f[2]=q0.z; f[3]=q0.w;
                f[4]=q1.x; f[5]=q1.y; f[6]=q1.z; f[7]=q1.w;
                f[8]=q2.x; f[9]=q2.y; f[10]=q2.z; f[11]=q2.w;
                f[12]=q3.x; f[13]=q3.y; f[14]=q3.z; f[15]=q3.w;
            } else {
                #pragma unroll
                for (int i = 0; i < 16; ++i) f[i] = 0.0f;
            }
            v8h h0, h1;
            #pragma unroll
            for (int i = 0; i < 8; ++i) { h0[i] = (_Float16)f[i]; h1[i] = (_Float16)f[8 + i]; }
            aDst[0] = h0;
            aDst[1] = h1;
        }
        // ---- stage B: coalesced float2 from W row, transposed into LDS ----
        {
            float w0 = 0.0f, w1 = 0.0f;
            if (gn + 1 < FDIM) {
                float2 q = *reinterpret_cast<const float2*>(W + (long long)(k0 + bk) * FDIM + gn);
                w0 = q.x; w1 = q.y;
            } else if (gn < FDIM) {
                w0 = W[(long long)(k0 + bk) * FDIM + gn];
            }
            lB[bn * 32 + bk]       = (_Float16)w0;
            lB[(bn + 1) * 32 + bk] = (_Float16)w1;
        }
        __syncthreads();
        acc = wmma_step(lA, lB, wave, lane, acc);
    }

    // ---- tail chunk k0 = 256, K valid = 12 ----
    {
        const int k0 = 256;
        __syncthreads();
        {
            v8h h0 = {}, h1 = {};
            if (rowOK && ak == 0) {
                #pragma unroll
                for (int i = 0; i < 8; ++i) h0[i] = (_Float16)xrow[k0 + i];
                #pragma unroll
                for (int i = 0; i < 4; ++i) h1[i] = (_Float16)xrow[k0 + 8 + i];
            }
            aDst[0] = h0;
            aDst[1] = h1;
        }
        {
            float w0 = 0.0f, w1 = 0.0f;
            const int gk = k0 + bk;
            if (gk < FDIM) {
                if (gn + 1 < FDIM) {
                    float2 q = *reinterpret_cast<const float2*>(W + (long long)gk * FDIM + gn);
                    w0 = q.x; w1 = q.y;
                } else if (gn < FDIM) {
                    w0 = W[(long long)gk * FDIM + gn];
                }
            }
            lB[bn * 32 + bk]       = (_Float16)w0;
            lB[(bn + 1) * 32 + bk] = (_Float16)w1;
        }
        __syncthreads();
        acc = wmma_step(lA, lB, wave, lane, acc);
    }

    // ---- store D: VGPR r -> M = r + 8*hi, N = lane&15 ----
    const int n  = lane & 15;
    const int hi = lane >> 4;
    const int col = n0 + n;
    if (col < FDIM) {
        #pragma unroll
        for (int r = 0; r < 8; ++r) {
            int grr = row0 + wave * 16 + r + hi * 8;
            if (grr < M) H[(long long)grr * FDIM + col] = acc[r];
        }
    }
}

// ---------------- pair aggregation: out1[col,:] += norm * H[row,:] ----------------
// One wave per (row,col,norm) pair; lanes move float4 chunks (67 per row).
__global__ void k_agg1(const int* __restrict__ rowI, const int* __restrict__ colI,
                       const float* __restrict__ dis, const float* __restrict__ H,
                       float* out1, int E, int N) {
    int gt   = blockIdx.x * blockDim.x + threadIdx.x;
    int pair = gt >> 5;
    int lane = gt & 31;
    int P = E + N;
    if (pair >= P) return;
    int r, c;
    if (pair < E) { r = rowI[pair]; c = colI[pair]; }
    else          { r = pair - E;   c = r; }
    float nrm = dis[r] * dis[c];
    const float4* hr = reinterpret_cast<const float4*>(H + (long long)r * FDIM);
    float* ob = out1 + (long long)c * FDIM;
    for (int j = lane; j < F4; j += 32) {
        float4 v = hr[j];
        int base = j * 4;
        atomicAdd(&ob[base + 0], nrm * v.x);
        atomicAdd(&ob[base + 1], nrm * v.y);
        atomicAdd(&ob[base + 2], nrm * v.z);
        atomicAdd(&ob[base + 3], nrm * v.w);
    }
}

// ---------------- H2 = relu(bn1(relu(out1 + b1))) ----------------
__global__ void k_elem1(const float* __restrict__ out1, const float* __restrict__ b1,
                        const float* __restrict__ g1, const float* __restrict__ beta1,
                        const float* __restrict__ rm1, const float* __restrict__ rv1,
                        float* H2, long long NH) {
    long long i = (long long)blockIdx.x * blockDim.x + threadIdx.x;
    if (i >= NH) return;
    int c = (int)(i % FDIM);
    float v = out1[i] + b1[c];
    v = fmaxf(v, 0.0f);
    v = (v - rm1[c]) * rsqrtf(rv1[c] + EPS) * g1[c] + beta1[c];
    H2[i] = fmaxf(v, 0.0f);
}

// ---------------- h2 = H2 @ W2  (one wave per node, shuffle reduce) ----------------
__global__ void k_dot(const float* __restrict__ H2, const float* __restrict__ W2,
                      float* h2, int N) {
    int gt   = blockIdx.x * blockDim.x + threadIdx.x;
    int node = gt >> 5;
    int lane = gt & 31;
    if (node >= N) return;
    const float* row = H2 + (long long)node * FDIM;
    float s = 0.0f;
    for (int j = lane; j < FDIM; j += 32) s += row[j] * W2[j];
    #pragma unroll
    for (int off = 16; off > 0; off >>= 1) s += __shfl_xor(s, off, 32);
    if (lane == 0) h2[node] = s;
}

// ---------------- scalar pair aggregation for layer 2 ----------------
__global__ void k_agg2(const int* __restrict__ rowI, const int* __restrict__ colI,
                       const float* __restrict__ dis, const float* __restrict__ h2,
                       float* out2, int E, int N) {
    int p = blockIdx.x * blockDim.x + threadIdx.x;
    int P = E + N;
    if (p >= P) return;
    int r, c;
    if (p < E) { r = rowI[p]; c = colI[p]; }
    else       { r = p - E;   c = r; }
    atomicAdd(&out2[c], dis[r] * dis[c] * h2[r]);
}

// ---------------- out = sigmoid(relu(bn2(out2 + b2))) ----------------
__global__ void k_final(const float* __restrict__ out2, const float* __restrict__ b2,
                        const float* __restrict__ g2, const float* __restrict__ beta2,
                        const float* __restrict__ rm2, const float* __restrict__ rv2,
                        float* out, int N) {
    int i = blockIdx.x * blockDim.x + threadIdx.x;
    if (i >= N) return;
    float v = out2[i] + b2[0];
    v = (v - rm2[0]) * rsqrtf(rv2[0] + EPS) * g2[0] + beta2[0];
    v = fmaxf(v, 0.0f);
    out[i] = 1.0f / (1.0f + expf(-v));
}

extern "C" void kernel_launch(void* const* d_in, const int* in_sizes, int n_in,
                              void* d_out, int out_size, void* d_ws, size_t ws_size,
                              hipStream_t stream) {
    const float* x     = (const float*)d_in[0];
    const int*   ei    = (const int*)  d_in[1];
    const float* W1    = (const float*)d_in[2];
    const float* b1    = (const float*)d_in[3];
    const float* g1    = (const float*)d_in[4];
    const float* beta1 = (const float*)d_in[5];
    const float* rm1   = (const float*)d_in[6];
    const float* rv1   = (const float*)d_in[7];
    const float* W2    = (const float*)d_in[8];
    const float* b2    = (const float*)d_in[9];
    const float* g2    = (const float*)d_in[10];
    const float* beta2 = (const float*)d_in[11];
    const float* rm2   = (const float*)d_in[12];
    const float* rv2   = (const float*)d_in[13];
    float* out = (float*)d_out;

    const int N = in_sizes[0] / FDIM;       // 20000
    const int E = in_sizes[1] / 2;          // 300000
    const int* rowI = ei;                   // edge_index[0]
    const int* colI = ei + E;               // edge_index[1]
    const long long NH = (long long)N * FDIM;

    // workspace layout (floats)
    float* ws   = (float*)d_ws;
    float* deg  = ws;                       // N
    float* dis  = ws + N;                   // N
    float* H    = ws + 2 * (long long)N;    // N*FDIM  (reused as H2)
    float* out1 = H + NH;                   // N*FDIM
    float* h2v  = out1 + NH;                // N
    float* out2 = h2v + N;                  // N
    float* H2   = H;

    const int P = E + N;                    // pairs incl. self-loops

    k_init<<<(int)((NH + 255) / 256), 256, 0, stream>>>(deg, out1, out2, N, NH);
    k_degree<<<(E + 255) / 256, 256, 0, stream>>>(colI, deg, E);
    k_dis<<<(N + 255) / 256, 256, 0, stream>>>(deg, dis, N);

    dim3 gGemm((FDIM + 15) / 16, (N + 127) / 128);
    k_gemm1<<<gGemm, 256, 0, stream>>>(x, W1, H, N);

    k_agg1<<<(P + 7) / 8, 256, 0, stream>>>(rowI, colI, dis, H, out1, E, N);
    k_elem1<<<(int)((NH + 255) / 256), 256, 0, stream>>>(out1, b1, g1, beta1, rm1, rv1, H2, NH);
    k_dot<<<(N + 7) / 8, 256, 0, stream>>>(H2, W2, h2v, N);
    k_agg2<<<(P + 255) / 256, 256, 0, stream>>>(rowI, colI, dis, h2v, out2, E, N);
    k_final<<<(N + 255) / 256, 256, 0, stream>>>(out2, b2, g2, beta2, rm2, rv2, out, N);
}
